// Model_shared_25769804008
// MI455X (gfx1250) — matrix operand order repaired and checked by reading, not compiled, not run on hardware
//
#include <hip/hip_runtime.h>
#include <hip/hip_bf16.h>

// ---------------------------------------------------------------------------
// CDNA5 (gfx1250) implementation of the GNN reference.
// All matmuls run on V_WMMA_F32_16X16X32_F16 (f16 inputs, f32 accumulate),
// 16 rows (nodes/edges) per wave32 tile. LDS used to re-layout between the
// WMMA D-layout (col-per-lane) and A-layout (row-per-lane). The encoder
// double-buffers its per-timestep inputs in LDS with
// GLOBAL_LOAD_ASYNC_TO_LDS_B128 + s_wait_asynccnt.
// ---------------------------------------------------------------------------

typedef __attribute__((ext_vector_type(16))) _Float16 v16h;
typedef __attribute__((ext_vector_type(8)))  _Float16 v8h;
typedef __attribute__((ext_vector_type(8)))  float    v8f;

static __device__ inline v16h pack16(v8h lo, v8h hi) {
    v16h r;
#pragma unroll
    for (int i = 0; i < 8; ++i) { r[i] = lo[i]; r[i + 8] = hi[i]; }
    return r;
}

// A-fragment (16x32 f16, M x K): lane = (h = lane>>4, n = lane&15), row = n.
// p points at this lane's row start (at k-base). ISA 7.12.2:
//  elements 0..7  -> K = h*8 + 0..7
//  elements 8..15 -> K = 16 + h*8 + 0..7
static __device__ inline v16h fragA(const _Float16* p, int h) {
    v8h lo = *(const v8h*)(p + h * 8);
    v8h hi = *(const v8h*)(p + 16 + h * 8);
    return pack16(lo, hi);
}

// B-fragment (32x16 f16, K x N) for computing X @ W^T from row-major W[out][in]:
// lane n = output column, kh = lane>>4; elements 0..7 -> K = kh*8 + 0..7, 8..15 -> K = 16 + kh*8 + 0..7.
static __device__ inline v16h fragB(const _Float16* W, int ld, int ntile, int kb, int lane) {
    int n = lane & 15, kh = lane >> 4;
    const _Float16* p = W + (size_t)(ntile * 16 + n) * ld + kb + kh * 8;
    v8h lo = *(const v8h*)p;
    v8h hi = *(const v8h*)(p + 16);
    return pack16(lo, hi);
}

static __device__ inline v8f wmma32(v16h a, v16h b, v8f c) {
    return __builtin_amdgcn_wmma_f32_16x16x32_f16(false, a, false, b, (short)0, c, false, false);
}

// fast math: v_rcp_f32 instead of IEEE division expansion
static __device__ inline float rcpf(float x) { return __builtin_amdgcn_rcpf(x); }
static __device__ inline float sigf(float x) { return rcpf(1.f + __expf(-x)); }
static __device__ inline float tanh_fast(float x) {
    float xc = fminf(fmaxf(x, -15.f), 15.f);
    float t = __expf(2.f * xc);
    return (t - 1.f) * rcpf(t + 1.f);
}

// --- gfx1250 async global->LDS copy (ASYNCcnt tracked) ----------------------
static __device__ inline void async_to_lds_b128(unsigned lds_off, const _Float16* gptr) {
    asm volatile("global_load_async_to_lds_b128 %0, %1, off"
                 :: "v"(lds_off), "v"((unsigned long long)(__SIZE_TYPE__)gptr)
                 : "memory");
}
static __device__ inline void wait_async_le2() {
    asm volatile("s_wait_asynccnt 0x2" ::: "memory");
}
static __device__ inline void wait_async_0() {
    asm volatile("s_wait_asynccnt 0x0" ::: "memory");
}

// ---------------------------------------------------------------------------
// small conversion / table kernels
// ---------------------------------------------------------------------------
__global__ void k_cvt(const float* __restrict__ src, _Float16* __restrict__ dst, long cnt) {
    long i = (long)blockIdx.x * blockDim.x + threadIdx.x;
    if (i < cnt) dst[i] = (_Float16)src[i];
}

__global__ void k_pe(_Float16* __restrict__ dpe) {  // posenc table [32][128]
    int i = blockIdx.x * blockDim.x + threadIdx.x;
    if (i >= 32 * 64) return;
    int p = i >> 6, ii = i & 63;
    float inv = __expf(-logf(10000.f) * (2.f * (float)ii / 128.f));
    float a = (float)p * inv;
    dpe[p * 128 + 2 * ii]     = (_Float16)sinf(a);
    dpe[p * 128 + 2 * ii + 1] = (_Float16)cosf(a);
}

// Ct[j][i] = sum_n Wk[n][j] * Wq[n][i]  (so qW = emb @ Ct^T, score = kf . qW[dst])
__global__ void k_ct(const float* __restrict__ Wq, const float* __restrict__ Wk,
                     _Float16* __restrict__ dCt) {
    int o = blockIdx.x * blockDim.x + threadIdx.x;
    if (o >= 128 * 128) return;
    int j = o >> 7, i2 = o & 127;
    float s = 0.f;
    for (int n = 0; n < 128; ++n) s += Wk[n * 128 + j] * Wq[n * 128 + i2];
    dCt[o] = (_Float16)s;
}

// edge_feat[e][:] = posenc[edge_type[e]] + op_emb[argmax(x[dst[e]])]
__global__ void k_edge_feat(const int* __restrict__ etype, const int* __restrict__ ei,
                            const float* __restrict__ x, const _Float16* __restrict__ dpe,
                            const _Float16* __restrict__ dOp, _Float16* __restrict__ ef,
                            int E) {
    int tid = blockIdx.x * blockDim.x + threadIdx.x;
    if (tid >= E * 8) return;
    int e = tid >> 3, c0 = (tid & 7) * 16;
    int et = etype[e];
    int d = ei[E + e];
    const float* xr = x + (size_t)d * 16;
    int op = 0; float best = xr[0];
#pragma unroll
    for (int k = 1; k < 16; ++k) { float v = xr[k]; if (v > best) { best = v; op = k; } }
#pragma unroll
    for (int c = 0; c < 16; ++c)
        ef[(size_t)e * 128 + c0 + c] = dpe[et * 128 + c0 + c] + dOp[op * 128 + c0 + c];
}

__global__ void k_zero(unsigned* __restrict__ segmax, float* __restrict__ den,
                       float* __restrict__ aggr, int N) {
    int t = blockIdx.x * blockDim.x + threadIdx.x;
    if (t < N) segmax[t] = 0u;
    else if (t < 2 * N) den[t - N] = 0.f;
    else if (t < 2 * N + 128 * N) aggr[t - 2 * N] = 0.f;
}

// ---------------------------------------------------------------------------
// GRU encoder: 2-layer GRU over T=32, one 16-node tile per wave.
// hbuf is [16 rows][64 cols] f16 in LDS (wave-private).
// ---------------------------------------------------------------------------
static __device__ inline void gru_layer_16x64(
    const v16h* aXf, int nkx, int ldx, const _Float16* __restrict__ Wih,
    const v16h* aHf, const _Float16* __restrict__ Whh,
    const float* __restrict__ bih, const float* __restrict__ bhh,
    _Float16* hbuf, int lane) {
    int n = lane & 15, h = lane >> 4;
    for (int j = 0; j < 4; ++j) {
        v8f accR = {}, accZ = {}, accNi = {}, accNh = {};
        for (int kt = 0; kt < nkx; ++kt) {
            accR  = wmma32(aXf[kt], fragB(Wih, ldx, j,     kt * 32, lane), accR);
            accZ  = wmma32(aXf[kt], fragB(Wih, ldx, 4 + j, kt * 32, lane), accZ);
            accNi = wmma32(aXf[kt], fragB(Wih, ldx, 8 + j, kt * 32, lane), accNi);
        }
#pragma unroll
        for (int kt = 0; kt < 2; ++kt) {
            accR  = wmma32(aHf[kt], fragB(Whh, 64, j,     kt * 32, lane), accR);
            accZ  = wmma32(aHf[kt], fragB(Whh, 64, 4 + j, kt * 32, lane), accZ);
            accNh = wmma32(aHf[kt], fragB(Whh, 64, 8 + j, kt * 32, lane), accNh);
        }
        int c = j * 16 + n;
        float br  = bih[c] + bhh[c];
        float bz  = bih[64 + c] + bhh[64 + c];
        float bni = bih[128 + c], bnh = bhh[128 + c];
#pragma unroll
        for (int r = 0; r < 8; ++r) {
            int m = r + 8 * h;                       // D-layout row
            float rr = sigf(accR[r] + br);
            float zz = sigf(accZ[r] + bz);
            float nn = tanh_fast(accNi[r] + bni + rr * (accNh[r] + bnh));
            float hold = (float)hbuf[m * 64 + c];
            hbuf[m * 64 + c] = (_Float16)((1.f - zz) * nn + zz * hold);
        }
    }
}

__global__ __launch_bounds__(128) void k_encoder(
    const _Float16* __restrict__ simres_h,
    const _Float16* __restrict__ Wih0, const _Float16* __restrict__ Whh0,
    const _Float16* __restrict__ Wih1, const _Float16* __restrict__ Whh1,
    const float* __restrict__ bih0, const float* __restrict__ bhh0,
    const float* __restrict__ bih1, const float* __restrict__ bhh1,
    const float* __restrict__ xin, float* __restrict__ out,
    _Float16* __restrict__ emb_h, int N) {
    int lane = threadIdx.x & 31, wid = threadIdx.x >> 5;
    int tile0 = (blockIdx.x * 4 + wid) * 16;
    if (tile0 >= N) return;
    int n = lane & 15, h = lane >> 4;

    __shared__ __align__(16) _Float16 sH[4][2][1024];
    __shared__ __align__(16) _Float16 sXs[4][2][512];   // 16 rows x 32 halves, x2 buffers
    _Float16* h0 = sH[wid][0];
    _Float16* h1 = sH[wid][1];
    for (int i = lane; i < 1024; i += 32) { h0[i] = (_Float16)0.f; h1[i] = (_Float16)0.f; }

    // async stage: lane (r = lane>>1, half = lane&1) copies 32 bytes of row r
    int arow = tile0 + (lane >> 1); if (arow >= N) arow = N - 1;
    const _Float16* agbase = simres_h + (size_t)arow * 1024 + (lane & 1) * 16;
    unsigned albase = (unsigned)(__SIZE_TYPE__)(void*)sXs[wid][0] +
                      (unsigned)((lane >> 1) * 64 + (lane & 1) * 32);
    const unsigned bufStride = 512 * 2;  // bytes between the two buffers

    // prime t = 0
    async_to_lds_b128(albase, agbase);
    async_to_lds_b128(albase + 16, agbase + 8);

    for (int t = 0; t < 32; ++t) {
        const _Float16* cur = sXs[wid][t & 1];
        if (t + 1 < 32) {  // issue next stage before consuming current
            unsigned dst = albase + ((t + 1) & 1) * bufStride;
            const _Float16* g = agbase + (t + 1) * 32;
            async_to_lds_b128(dst, g);
            async_to_lds_b128(dst + 16, g + 8);
            wait_async_le2();   // in-order completion: stage t is resident
        } else {
            wait_async_0();
        }
        v16h aX[1]; aX[0] = fragA(cur + n * 32, h);
        v16h aH0[2] = { fragA(h0 + n * 64, h), fragA(h0 + n * 64 + 32, h) };
        gru_layer_16x64(aX, 1, 32, Wih0, aH0, Whh0, bih0, bhh0, h0, lane);
        v16h aX1[2] = { fragA(h0 + n * 64, h), fragA(h0 + n * 64 + 32, h) };
        v16h aH1[2] = { fragA(h1 + n * 64, h), fragA(h1 + n * 64 + 32, h) };
        gru_layer_16x64(aX1, 2, 64, Wih1, aH1, Whh1, bih1, bhh1, h1, lane);
    }
    // enc = concat(h0,h1), masked by node type (Input|Const)
    for (int i = lane; i < 1024; i += 32) {
        int m = i >> 6, c = i & 63;
        int node = tile0 + m;
        if (node < N) {
            bool msk = (xin[(size_t)node * 16] == 1.f) || (xin[(size_t)node * 16 + 1] == 1.f);
            float v0 = msk ? (float)h0[i] : 0.f;
            float v1 = msk ? (float)h1[i] : 0.f;
            out[(size_t)node * 128 + c]      = v0;
            out[(size_t)node * 128 + 64 + c] = v1;
            emb_h[(size_t)node * 128 + c]      = (_Float16)v0;
            emb_h[(size_t)node * 128 + 64 + c] = (_Float16)v1;
        }
    }
}

// qW = emb @ Ct^T  (folded Wq^T Wk), per 16-node tile
__global__ __launch_bounds__(128) void k_qnode(
    const _Float16* __restrict__ emb_h, const _Float16* __restrict__ Ct,
    _Float16* __restrict__ qW, int N) {
    int lane = threadIdx.x & 31, wid = threadIdx.x >> 5;
    int tile0 = (blockIdx.x * 4 + wid) * 16;
    if (tile0 >= N) return;
    int n = lane & 15, h = lane >> 4;
    int row = tile0 + n; if (row >= N) row = N - 1;
    v16h aE[4];
#pragma unroll
    for (int kt = 0; kt < 4; ++kt) aE[kt] = fragA(emb_h + (size_t)row * 128 + kt * 32, h);
    for (int j = 0; j < 8; ++j) {
        v8f acc = {};
#pragma unroll
        for (int kt = 0; kt < 4; ++kt) acc = wmma32(aE[kt], fragB(Ct, 128, j, kt * 32, lane), acc);
#pragma unroll
        for (int r = 0; r < 8; ++r) {
            int node = tile0 + r + 8 * h;
            if (node < N) qW[(size_t)node * 128 + j * 16 + n] = (_Float16)acc[r];
        }
    }
}

// Per 16-edge tile: kf = emb[src] + edge_feat; score = kf.qW[dst]; v = kf @ Wv^T
__global__ __launch_bounds__(128) void k_edge(
    const _Float16* __restrict__ emb_h, const _Float16* __restrict__ ef,
    const _Float16* __restrict__ qW, const _Float16* __restrict__ Wv,
    const int* __restrict__ ei, float* __restrict__ score,
    unsigned* __restrict__ segmax, _Float16* __restrict__ vbuf, int E) {
    int lane = threadIdx.x & 31, wid = threadIdx.x >> 5;
    int tile0 = (blockIdx.x * 4 + wid) * 16;
    if (tile0 >= E) return;
    int n = lane & 15, h = lane >> 4;
    int e = tile0 + n; if (e >= E) e = E - 1;
    int s = ei[e], d = ei[E + e];
    const _Float16* ep = emb_h + (size_t)s * 128;
    const _Float16* fp = ef + (size_t)e * 128;
    const _Float16* qp = qW + (size_t)d * 128;
    float dot = 0.f;
    v16h kf[4];
#pragma unroll
    for (int kt = 0; kt < 4; ++kt) {
        int kb = kt * 32;
        v8h alo = *(const v8h*)(ep + kb + h * 8);
        v8h ahi = *(const v8h*)(ep + kb + 16 + h * 8);
        v8h flo = *(const v8h*)(fp + kb + h * 8);
        v8h fhi = *(const v8h*)(fp + kb + 16 + h * 8);
        v8h qlo = *(const v8h*)(qp + kb + h * 8);
        v8h qhi = *(const v8h*)(qp + kb + 16 + h * 8);
        v8h slo = alo + flo, shi = ahi + fhi;
        kf[kt] = pack16(slo, shi);
#pragma unroll
        for (int i = 0; i < 8; ++i)
            dot += (float)slo[i] * (float)qlo[i] + (float)shi[i] * (float)qhi[i];
    }
    dot += __shfl_xor(dot, 16, 32);  // combine the two K-halves of each edge row
    if (h == 0 && tile0 + n < E) {
        float sc = dot * 0.0883883476f;  // 1/sqrt(128)
        score[e] = sc;
        unsigned u = __float_as_uint(sc);
        unsigned enc = (u & 0x80000000u) ? ~u : (u | 0x80000000u);
        atomicMax(segmax + d, enc);
    }
    for (int j = 0; j < 8; ++j) {
        v8f acc = {};
#pragma unroll
        for (int kt = 0; kt < 4; ++kt) acc = wmma32(kf[kt], fragB(Wv, 128, j, kt * 32, lane), acc);
#pragma unroll
        for (int r = 0; r < 8; ++r) {
            int er = tile0 + r + 8 * h;
            if (er < E) vbuf[(size_t)er * 128 + j * 16 + n] = (_Float16)acc[r];
        }
    }
}

__global__ void k_exden(const int* __restrict__ ei, const float* __restrict__ score,
                        const unsigned* __restrict__ segmax, float* __restrict__ ex,
                        float* __restrict__ den, int E) {
    int e = blockIdx.x * blockDim.x + threadIdx.x;
    if (e >= E) return;
    int d = ei[E + e];
    unsigned u = segmax[d];
    float mx = __uint_as_float((u & 0x80000000u) ? (u & 0x7FFFFFFFu) : ~u);
    float v = __expf(score[e] - mx);
    ex[e] = v;
    atomicAdd(den + d, v);
}

__global__ void k_aggr(const int* __restrict__ ei, const float* __restrict__ ex,
                       const float* __restrict__ den, const _Float16* __restrict__ vbuf,
                       float* __restrict__ aggr, int E) {
    int t = blockIdx.x * blockDim.x + threadIdx.x;
    int e = t >> 2;
    if (e >= E) return;
    int part = t & 3;
    int d = ei[E + e];
    float alpha = ex[e] * rcpf(den[d] + 1e-9f);
    const _Float16* vp = vbuf + (size_t)e * 128 + part * 32;
    float* ap = aggr + (size_t)d * 128 + part * 32;
#pragma unroll
    for (int c = 0; c < 32; ++c) atomicAdd(ap + c, alpha * (float)vp[c]);
}

// Per 16-node tile: MLP(relu) -> MLP -> GRU(node_emb), all on WMMA.
__global__ __launch_bounds__(128) void k_update(
    const float* __restrict__ aggr,
    const _Float16* __restrict__ W1h, const float* __restrict__ b1,
    const _Float16* __restrict__ W2h, const float* __restrict__ b2,
    const _Float16* __restrict__ gWih, const _Float16* __restrict__ gWhh,
    const float* __restrict__ gbih, const float* __restrict__ gbhh,
    float* __restrict__ out, _Float16* __restrict__ emb_h, int N) {
    int lane = threadIdx.x & 31, wid = threadIdx.x >> 5;
    int tile0 = (blockIdx.x * 4 + wid) * 16;
    if (tile0 >= N) return;
    int n = lane & 15, h = lane >> 4;
    int row = tile0 + n; if (row >= N) row = N - 1;

    __shared__ __align__(16) _Float16 sbuf[4][2][2048];
    _Float16* sA = sbuf[wid][0];
    _Float16* sB = sbuf[wid][1];

    // aggr (f32) -> A fragments (f16)
    v16h aG[4];
#pragma unroll
    for (int kt = 0; kt < 4; ++kt) {
        const float* p = aggr + (size_t)row * 128 + kt * 32;
        v16h a;
#pragma unroll
        for (int i = 0; i < 8; ++i) {
            a[i]     = (_Float16)p[h * 8 + i];
            a[i + 8] = (_Float16)p[16 + h * 8 + i];
        }
        aG[kt] = a;
    }
    // mlp1 = relu(aggr @ W1^T + b1) -> sA
    for (int j = 0; j < 8; ++j) {
        v8f acc = {};
#pragma unroll
        for (int kt = 0; kt < 4; ++kt) acc = wmma32(aG[kt], fragB(W1h, 128, j, kt * 32, lane), acc);
        int c = j * 16 + n;
#pragma unroll
        for (int r = 0; r < 8; ++r) {
            int m = r + 8 * h;
            float v = acc[r] + b1[c];
            sA[m * 128 + c] = (_Float16)(v > 0.f ? v : 0.f);
        }
    }
    v16h aM[4];
#pragma unroll
    for (int kt = 0; kt < 4; ++kt) aM[kt] = fragA(sA + n * 128 + kt * 32, h);
    // mlp2 = mlp1 @ W2^T + b2 -> sB (GRU input x)
    for (int j = 0; j < 8; ++j) {
        v8f acc = {};
#pragma unroll
        for (int kt = 0; kt < 4; ++kt) acc = wmma32(aM[kt], fragB(W2h, 128, j, kt * 32, lane), acc);
        int c = j * 16 + n;
#pragma unroll
        for (int r = 0; r < 8; ++r) {
            int m = r + 8 * h;
            sB[m * 128 + c] = (_Float16)(acc[r] + b2[c]);
        }
    }
    v16h aXf[4], aHf[4];
#pragma unroll
    for (int kt = 0; kt < 4; ++kt) {
        aXf[kt] = fragA(sB + n * 128 + kt * 32, h);
        aHf[kt] = fragA(emb_h + (size_t)row * 128 + kt * 32, h);
    }
    // GRU: r,z,n over 3*128 gate columns (ntiles: r=j, z=8+j, n=16+j)
    for (int j = 0; j < 8; ++j) {
        v8f accR = {}, accZ = {}, accNi = {}, accNh = {};
#pragma unroll
        for (int kt = 0; kt < 4; ++kt) {
            int kb = kt * 32;
            accR  = wmma32(aXf[kt], fragB(gWih, 128, j,      kb, lane), accR);
            accZ  = wmma32(aXf[kt], fragB(gWih, 128, 8 + j,  kb, lane), accZ);
            accNi = wmma32(aXf[kt], fragB(gWih, 128, 16 + j, kb, lane), accNi);
            accR  = wmma32(aHf[kt], fragB(gWhh, 128, j,      kb, lane), accR);
            accZ  = wmma32(aHf[kt], fragB(gWhh, 128, 8 + j,  kb, lane), accZ);
            accNh = wmma32(aHf[kt], fragB(gWhh, 128, 16 + j, kb, lane), accNh);
        }
        int c = j * 16 + n;
        float br  = gbih[c] + gbhh[c];
        float bz  = gbih[128 + c] + gbhh[128 + c];
        float bni = gbih[256 + c], bnh = gbhh[256 + c];
#pragma unroll
        for (int r = 0; r < 8; ++r) {
            int node = tile0 + r + 8 * h;
            float rr = sigf(accR[r] + br);
            float zz = sigf(accZ[r] + bz);
            float nn = tanh_fast(accNi[r] + bni + rr * (accNh[r] + bnh));
            if (node < N) {
                float hold = out[(size_t)node * 128 + c];
                float hnew = (1.f - zz) * nn + zz * hold;
                out[(size_t)node * 128 + c] = hnew;
                emb_h[(size_t)node * 128 + c] = (_Float16)hnew;
            }
        }
    }
}

// ---------------------------------------------------------------------------
extern "C" void kernel_launch(void* const* d_in, const int* in_sizes, int n_in,
                              void* d_out, int out_size, void* d_ws, size_t ws_size,
                              hipStream_t stream) {
    const float* x      = (const float*)d_in[0];
    const int*   ei     = (const int*)d_in[1];
    const int*   etype  = (const int*)d_in[2];
    const float* simres = (const float*)d_in[3];
    const float* eWih0 = (const float*)d_in[5];
    const float* eWhh0 = (const float*)d_in[6];
    const float* ebih0 = (const float*)d_in[7];
    const float* ebhh0 = (const float*)d_in[8];
    const float* eWih1 = (const float*)d_in[9];
    const float* eWhh1 = (const float*)d_in[10];
    const float* ebih1 = (const float*)d_in[11];
    const float* ebhh1 = (const float*)d_in[12];
    const float* Wq    = (const float*)d_in[13];
    const float* Wk    = (const float*)d_in[14];
    const float* Wv    = (const float*)d_in[15];
    const float* opE   = (const float*)d_in[16];
    const float* W1    = (const float*)d_in[17];
    const float* b1    = (const float*)d_in[18];
    const float* W2    = (const float*)d_in[19];
    const float* b2    = (const float*)d_in[20];
    const float* gWih  = (const float*)d_in[21];
    const float* gWhh  = (const float*)d_in[22];
    const float* gbih  = (const float*)d_in[23];
    const float* gbhh  = (const float*)d_in[24];
    float* out = (float*)d_out;

    const int N = in_sizes[0] / 16;
    const int E = in_sizes[2];

    char* base = (char*)d_ws;
    size_t off = 0;
    auto take = [&](size_t bytes) -> void* {
        void* p = base + off;
        off += (bytes + 255) & ~(size_t)255;
        return p;
    };
    _Float16* simres_h = (_Float16*)take((size_t)N * 1024 * 2);
    _Float16* ef       = (_Float16*)take((size_t)E * 128 * 2);
    _Float16* vbuf     = (_Float16*)take((size_t)E * 128 * 2);
    _Float16* emb_h    = (_Float16*)take((size_t)N * 128 * 2);
    _Float16* qW       = (_Float16*)take((size_t)N * 128 * 2);
    float*    score    = (float*)take((size_t)E * 4);
    float*    exb      = (float*)take((size_t)E * 4);
    unsigned* segmax   = (unsigned*)take((size_t)N * 4);
    float*    den      = (float*)take((size_t)N * 4);
    float*    aggr     = (float*)take((size_t)N * 128 * 4);
    _Float16* dWih0 = (_Float16*)take(192 * 32 * 2);
    _Float16* dWhh0 = (_Float16*)take(192 * 64 * 2);
    _Float16* dWih1 = (_Float16*)take(192 * 64 * 2);
    _Float16* dWhh1 = (_Float16*)take(192 * 64 * 2);
    _Float16* dWv   = (_Float16*)take(128 * 128 * 2);
    _Float16* dW1   = (_Float16*)take(128 * 128 * 2);
    _Float16* dW2   = (_Float16*)take(128 * 128 * 2);
    _Float16* dgWih = (_Float16*)take(384 * 128 * 2);
    _Float16* dgWhh = (_Float16*)take(384 * 128 * 2);
    _Float16* dOp   = (_Float16*)take(16 * 128 * 2);
    _Float16* dCt   = (_Float16*)take(128 * 128 * 2);
    _Float16* dpe   = (_Float16*)take(32 * 128 * 2);

    auto cvt = [&](const float* s, _Float16* d, long cnt) {
        k_cvt<<<(unsigned)((cnt + 255) / 256), 256, 0, stream>>>(s, d, cnt);
    };
    cvt(simres, simres_h, (long)N * 1024);
    cvt(eWih0, dWih0, 192 * 32);
    cvt(eWhh0, dWhh0, 192 * 64);
    cvt(eWih1, dWih1, 192 * 64);
    cvt(eWhh1, dWhh1, 192 * 64);
    cvt(Wv, dWv, 128 * 128);
    cvt(W1, dW1, 128 * 128);
    cvt(W2, dW2, 128 * 128);
    cvt(gWih, dgWih, 384 * 128);
    cvt(gWhh, dgWhh, 384 * 128);
    cvt(opE, dOp, 16 * 128);
    k_pe<<<8, 256, 0, stream>>>(dpe);
    k_ct<<<64, 256, 0, stream>>>(Wq, Wk, dCt);
    k_edge_feat<<<(unsigned)((E * 8 + 255) / 256), 256, 0, stream>>>(etype, ei, x, dpe, dOp, ef, E);

    const int nTiles = (N + 15) / 16;
    const int eTiles = (E + 15) / 16;
    const unsigned nBlocks = (unsigned)((nTiles + 3) / 4);
    const unsigned eBlocks = (unsigned)((eTiles + 3) / 4);

    k_encoder<<<nBlocks, 128, 0, stream>>>(simres_h, dWih0, dWhh0, dWih1, dWhh1,
                                           ebih0, ebhh0, ebih1, ebhh1, x, out, emb_h, N);

    for (int round = 0; round < 2; ++round) {
        int zn = N * 130;
        k_zero<<<(unsigned)((zn + 255) / 256), 256, 0, stream>>>(segmax, den, aggr, N);
        k_qnode<<<nBlocks, 128, 0, stream>>>(emb_h, dCt, qW, N);
        k_edge<<<eBlocks, 128, 0, stream>>>(emb_h, ef, qW, dWv, ei, score, segmax, vbuf, E);
        k_exden<<<(unsigned)((E + 255) / 256), 256, 0, stream>>>(ei, score, segmax, exb, den, E);
        k_aggr<<<(unsigned)((E * 4 + 255) / 256), 256, 0, stream>>>(ei, exb, den, vbuf, aggr, E);
        k_update<<<nBlocks, 128, 0, stream>>>(aggr, dW1, b1, dW2, b2, dgWih, dgWhh,
                                              gbih, gbhh, out, emb_h, N);
    }
}
